// JKNetMaxpool_76922864272074
// MI455X (gfx1250) — matrix operand order repaired and checked by reading, not compiled, not run on hardware
//
#include <hip/hip_runtime.h>
#include <stdint.h>

// ---------------------------------------------------------------------------
// JK-Net (6x GCNConv + JK-max + normalized GCNConv + log_softmax) for MI455X.
// GEMMs run on v_wmma_f32_16x16x32_bf16 (fp32 accum), one wave per 16x32
// output strip (two N-tiles, shared A fragment). Edge aggregation uses native
// f32 global atomics (L2-resident dst buffer); everything else fused.
// M (=10000) is an exact multiple of 16, so the GEMM has no edge guards and
// EXEC is statically all-ones at every WMMA.
// ---------------------------------------------------------------------------

typedef __attribute__((ext_vector_type(16))) __bf16 v16bf;
typedef __attribute__((ext_vector_type(8)))  float  v8f;

#define NN   10000     // nodes (multiple of 16)
#define NE   320000    // edges
#define F_IN 512
#define F_H  256
#define F_O  64
#define N_HID 5        // hidden convs after layer 0

static __device__ __forceinline__ uint16_t f2bf(float f) {
  union { float f; uint32_t u; } c; c.f = f;
  uint32_t r = c.u + 0x7FFFu + ((c.u >> 16) & 1u);   // round-to-nearest-even
  return (uint16_t)(r >> 16);
}

__global__ void k_f32_to_bf16(const float* __restrict__ in,
                              uint16_t* __restrict__ out, int n) {
  int i = blockIdx.x * blockDim.x + threadIdx.x;
  if (i < n) out[i] = f2bf(in[i]);
}

__global__ void k_zero_f32(float* __restrict__ p, int n) {
  int i = blockIdx.x * blockDim.x + threadIdx.x;
  if (i < n) p[i] = 0.0f;
}

// Pack W[K,N] (fp32, row-major) into a bf16 WMMA B-fragment stream:
// stream index = ((nt*Kchunks + kt)*32 + lane)*16 + i, element = W[k][n] with
// n = nt*16 + (lane&15), k = kt*32 + (lane<16 ? 0 : 8) + (i<8 ? i : 8 + i)   (ISA 16-bit B layout)
__global__ void k_pack_w(const float* __restrict__ W,
                         uint16_t* __restrict__ out, int K, int N) {
  int tid = blockIdx.x * blockDim.x + threadIdx.x;
  if (tid >= K * N) return;
  int i    = tid & 15;
  int lane = (tid >> 4) & 31;
  int rest = tid >> 9;
  int kchunks = K >> 5;
  int kt = rest % kchunks;
  int nt = rest / kchunks;
  int n  = nt * 16 + (lane & 15);
  int k  = kt * 32 + ((lane < 16) ? 0 : 8) + ((i < 8) ? i : 8 + i);
  out[tid] = f2bf(W[k * N + n]);
}

// C[M,N] = A[M,K] x B[K,N]. A bf16 row-major, B pre-packed fragments.
// One wave per 16x32 strip of C: two accumulators share one A fragment.
// Requires M % 16 == 0, N % 32 == 0, K % 32 == 0.
__global__ void k_gemm_bf16_wmma(const uint16_t* __restrict__ A,
                                 const uint16_t* __restrict__ Bp,
                                 float* __restrict__ C,
                                 int M, int K, int N) {
  const int lane = threadIdx.x & 31;
  const int wave = blockIdx.x * (blockDim.x >> 5) + (threadIdx.x >> 5);
  const int nPairs = N >> 5;
  const int mTiles = M >> 4;
  if (wave >= mTiles * nPairs) return;     // wave-uniform: EXEC stays all-1
  const int mt = wave / nPairs;
  const int np = wave % nPairs;
  const int kchunks = K >> 5;

  // A fragment: lane holds row (lane&15); K chunks {b..b+7} and {b+16..b+23},
  // b = 0 (lanes 0-15) or 8 (lanes 16-31)  -> two contiguous 16B loads.
  const int aRow = mt * 16 + (lane & 15);
  const uint16_t* aPtr  = A + (size_t)aRow * K + ((lane < 16) ? 0 : 8);
  const uint16_t* b0Ptr = Bp + ((size_t)(2 * np)     * kchunks) * 512 + lane * 16;
  const uint16_t* b1Ptr = Bp + ((size_t)(2 * np + 1) * kchunks) * 512 + lane * 16;

  v8f c0 = {};
  v8f c1 = {};
  for (int kt = 0; kt < kchunks; ++kt) {
    union { uint4 u[2]; v16bf v; } a, b0, b1;
    a.u[0]  = *reinterpret_cast<const uint4*>(aPtr);
    a.u[1]  = *reinterpret_cast<const uint4*>(aPtr + 16);
    b0.u[0] = *reinterpret_cast<const uint4*>(b0Ptr);
    b0.u[1] = *reinterpret_cast<const uint4*>(b0Ptr + 8);
    b1.u[0] = *reinterpret_cast<const uint4*>(b1Ptr);
    b1.u[1] = *reinterpret_cast<const uint4*>(b1Ptr + 8);
    aPtr  += 32;
    b0Ptr += 512;
    b1Ptr += 512;
    __builtin_prefetch(aPtr,  0, 3);   // global_prefetch_b8
    __builtin_prefetch(b0Ptr, 0, 3);
    __builtin_prefetch(b1Ptr, 0, 3);
    c0 = __builtin_amdgcn_wmma_f32_16x16x32_bf16(
        false, a.v, false, b0.v, (short)0, c0, false, false);
    c1 = __builtin_amdgcn_wmma_f32_16x16x32_bf16(
        false, a.v, false, b1.v, (short)0, c1, false, false);
  }

  // C layout: VGPR i -> row = i + (lane<16 ? 0 : 8), col = lane&15
  const int col0  = np * 32 + (lane & 15);
  const int rBase = mt * 16 + ((lane < 16) ? 0 : 8);
#pragma unroll
  for (int i = 0; i < 8; ++i) {
    const size_t r = (size_t)(rBase + i) * N;
    C[r + col0]      = c0[i];
    C[r + col0 + 16] = c1[i];
  }
}

// agg[dst] += m[src] over edges; 4 features per thread (float4 gather).
__global__ void k_scatter_add(const float* __restrict__ m,
                              const long long* __restrict__ ei,
                              float* __restrict__ agg, int E, int D) {
  int idx = blockIdx.x * blockDim.x + threadIdx.x;
  int chunks = D >> 2;
  if (idx >= E * chunks) return;
  int e = idx / chunks;
  int c = (idx - e * chunks) << 2;
  int src = (int)ei[e];
  int dst = (int)ei[E + e];
  const float4 v = *reinterpret_cast<const float4*>(m + (size_t)src * D + c);
  float* p = agg + (size_t)dst * D + c;
  __hip_atomic_fetch_add(p + 0, v.x, __ATOMIC_RELAXED, __HIP_MEMORY_SCOPE_AGENT);
  __hip_atomic_fetch_add(p + 1, v.y, __ATOMIC_RELAXED, __HIP_MEMORY_SCOPE_AGENT);
  __hip_atomic_fetch_add(p + 2, v.z, __ATOMIC_RELAXED, __HIP_MEMORY_SCOPE_AGENT);
  __hip_atomic_fetch_add(p + 3, v.w, __ATOMIC_RELAXED, __HIP_MEMORY_SCOPE_AGENT);
}

// x = relu(agg + b); jk = max(jk, x); xbf = bf16(x)   (fused epilogue)
__global__ void k_bias_relu_max(const float* __restrict__ agg,
                                const float* __restrict__ b,
                                float* __restrict__ jk,
                                uint16_t* __restrict__ xbf, int total, int D) {
  int i = blockIdx.x * blockDim.x + threadIdx.x;
  if (i >= total) return;
  float x = agg[i] + b[i & (D - 1)];
  x = x > 0.0f ? x : 0.0f;
  float o = jk[i];
  jk[i] = o > x ? o : x;                 // jk pre-zeroed; relu >= 0 so valid
  xbf[i] = f2bf(x);
}

__global__ void k_deg(const long long* __restrict__ ei,
                      float* __restrict__ deg, int E) {
  int e = blockIdx.x * blockDim.x + threadIdx.x;
  if (e >= E) return;
  int dst = (int)ei[E + e];
  __hip_atomic_fetch_add(deg + dst, 1.0f, __ATOMIC_RELAXED, __HIP_MEMORY_SCOPE_AGENT);
}

// aggo[dst] += mo[src] * dinv[src]*dinv[dst], dinv = (deg+1)^-0.5, D = 64
__global__ void k_scatter_norm(const float* __restrict__ mo,
                               const long long* __restrict__ ei,
                               const float* __restrict__ deg,
                               float* __restrict__ aggo, int E) {
  int idx = blockIdx.x * blockDim.x + threadIdx.x;
  if (idx >= E * 16) return;
  int e = idx >> 4;
  int c = (idx & 15) << 2;
  int src = (int)ei[e];
  int dst = (int)ei[E + e];
  float coef = rsqrtf(deg[src] + 1.0f) * rsqrtf(deg[dst] + 1.0f);
  const float4 v = *reinterpret_cast<const float4*>(mo + (size_t)src * F_O + c);
  float* p = aggo + (size_t)dst * F_O + c;
  __hip_atomic_fetch_add(p + 0, v.x * coef, __ATOMIC_RELAXED, __HIP_MEMORY_SCOPE_AGENT);
  __hip_atomic_fetch_add(p + 1, v.y * coef, __ATOMIC_RELAXED, __HIP_MEMORY_SCOPE_AGENT);
  __hip_atomic_fetch_add(p + 2, v.z * coef, __ATOMIC_RELAXED, __HIP_MEMORY_SCOPE_AGENT);
  __hip_atomic_fetch_add(p + 3, v.w * coef, __ATOMIC_RELAXED, __HIP_MEMORY_SCOPE_AGENT);
}

// out = log_softmax(aggo + mo/(deg+1) + bo). One wave per row, 2 feats/lane.
__global__ void k_final(const float* __restrict__ aggo,
                        const float* __restrict__ mo,
                        const float* __restrict__ deg,
                        const float* __restrict__ bo,
                        float* __restrict__ out, int n) {
  int lane = threadIdx.x & 31;
  int row  = blockIdx.x * (blockDim.x >> 5) + (threadIdx.x >> 5);
  if (row >= n) return;
  float inv = 1.0f / (deg[row] + 1.0f);
  size_t base = (size_t)row * F_O;
  float v0 = aggo[base + lane]      + mo[base + lane]      * inv + bo[lane];
  float v1 = aggo[base + lane + 32] + mo[base + lane + 32] * inv + bo[lane + 32];
  float mx = fmaxf(v0, v1);
  for (int m = 16; m; m >>= 1) mx = fmaxf(mx, __shfl_xor(mx, m, 32));
  float s = __expf(v0 - mx) + __expf(v1 - mx);
  for (int m = 16; m; m >>= 1) s += __shfl_xor(s, m, 32);
  float lse = mx + __logf(s);
  out[base + lane]      = v0 - lse;
  out[base + lane + 32] = v1 - lse;
}

// ---------------------------------------------------------------------------

extern "C" void kernel_launch(void* const* d_in, const int* in_sizes, int n_in,
                              void* d_out, int out_size, void* d_ws, size_t ws_size,
                              hipStream_t stream) {
  (void)in_sizes; (void)n_in; (void)out_size; (void)ws_size;

  const float*     h  = (const float*)d_in[0];
  const long long* ei = (const long long*)d_in[1];
  const float*     W0 = (const float*)d_in[2];
  const float*     b0 = (const float*)d_in[3];
  const float*     Wh = (const float*)d_in[4];
  const float*     bh = (const float*)d_in[5];
  const float*     Wo = (const float*)d_in[6];
  const float*     bo = (const float*)d_in[7];
  float* out = (float*)d_out;

  char* ws = (char*)d_ws;
  size_t off = 0;
  auto alloc = [&](size_t bytes) {
    void* p = ws + off;
    off = (off + bytes + 255) & ~(size_t)255;
    return p;
  };
  uint16_t* hbf  = (uint16_t*)alloc((size_t)NN * F_IN * 2);
  uint16_t* w0p  = (uint16_t*)alloc((size_t)F_IN * F_H * 2);
  uint16_t* whp  = (uint16_t*)alloc((size_t)N_HID * F_H * F_H * 2);
  uint16_t* wop  = (uint16_t*)alloc((size_t)F_H * F_O * 2);
  float*    m    = (float*)   alloc((size_t)NN * F_H * 4);
  float*    agg  = (float*)   alloc((size_t)NN * F_H * 4);
  uint16_t* xbf  = (uint16_t*)alloc((size_t)NN * F_H * 2);
  float*    jk   = (float*)   alloc((size_t)NN * F_H * 4);
  uint16_t* jkbf = (uint16_t*)alloc((size_t)NN * F_H * 2);
  float*    deg  = (float*)   alloc((size_t)NN * 4);
  float*    mo   = (float*)   alloc((size_t)NN * F_O * 4);
  float*    aggo = (float*)   alloc((size_t)NN * F_O * 4);

  const int T = 256;
  auto blocks = [](int n, int t) { return (n + t - 1) / t; };
  auto gemm = [&](const uint16_t* A, const uint16_t* Bp, float* C,
                  int M, int K, int N) {
    int waves = (M >> 4) * (N >> 5);      // one wave per 16x32 output strip
    k_gemm_bf16_wmma<<<blocks(waves * 32, T), T, 0, stream>>>(A, Bp, C, M, K, N);
  };

  // ---- one-time conversions / weight packing ----
  k_f32_to_bf16<<<blocks(NN * F_IN, T), T, 0, stream>>>(h, hbf, NN * F_IN);
  k_pack_w<<<blocks(F_IN * F_H, T), T, 0, stream>>>(W0, w0p, F_IN, F_H);
  for (int i = 0; i < N_HID; ++i)
    k_pack_w<<<blocks(F_H * F_H, T), T, 0, stream>>>(
        Wh + (size_t)i * F_H * F_H, whp + (size_t)i * F_H * F_H, F_H, F_H);
  k_pack_w<<<blocks(F_H * F_O, T), T, 0, stream>>>(Wo, wop, F_H, F_O);
  k_zero_f32<<<blocks(NN * F_H, T), T, 0, stream>>>(jk, NN * F_H);

  // ---- layer 0 ----
  gemm(hbf, w0p, m, NN, F_IN, F_H);
  k_zero_f32<<<blocks(NN * F_H, T), T, 0, stream>>>(agg, NN * F_H);
  k_scatter_add<<<blocks(NE * (F_H / 4), T), T, 0, stream>>>(m, ei, agg, NE, F_H);
  k_bias_relu_max<<<blocks(NN * F_H, T), T, 0, stream>>>(agg, b0, jk, xbf, NN * F_H, F_H);

  // ---- hidden layers ----
  for (int i = 0; i < N_HID; ++i) {
    gemm(xbf, whp + (size_t)i * F_H * F_H, m, NN, F_H, F_H);
    k_zero_f32<<<blocks(NN * F_H, T), T, 0, stream>>>(agg, NN * F_H);
    k_scatter_add<<<blocks(NE * (F_H / 4), T), T, 0, stream>>>(m, ei, agg, NE, F_H);
    k_bias_relu_max<<<blocks(NN * F_H, T), T, 0, stream>>>(
        agg, bh + (size_t)i * F_H, jk, xbf, NN * F_H, F_H);
  }

  // ---- final normalized GCNConv + log_softmax ----
  k_f32_to_bf16<<<blocks(NN * F_H, T), T, 0, stream>>>(jk, jkbf, NN * F_H);
  gemm(jkbf, wop, mo, NN, F_H, F_O);
  k_zero_f32<<<blocks(NN, T), T, 0, stream>>>(deg, NN);
  k_deg<<<blocks(NE, T), T, 0, stream>>>(ei, deg, NE);
  k_zero_f32<<<blocks(NN * F_O, T), T, 0, stream>>>(aggo, NN * F_O);
  k_scatter_norm<<<blocks(NE * 16, T), T, 0, stream>>>(mo, ei, deg, aggo, NE);
  k_final<<<blocks(NN * 32, T), T, 0, stream>>>(aggo, mo, deg, bo, out, NN);
}
